// SpatioTemporalAutoEncoder_47321949667763
// MI455X (gfx1250) — matrix operand order repaired and checked by reading, not compiled, not run on hardware
//
#include <hip/hip_runtime.h>
#include <hip/hip_bf16.h>
#include <stdint.h>

// ---------------------------------------------------------------------------
// CDNA5 (gfx1250) implicit-GEMM conv engine.
//   - bf16 WMMA (v_wmma_f32_16x16x32_bf16), f32 accumulate
//   - weights pre-converted to bf16 once; 32K x 64N bf16 tile staged via
//     async-to-LDS (GVS form: SGPR base + 32-bit offset), B fragments are
//     raw ds_load_u16 (no per-iteration converts)
//   - 128-thread blocks (4 waves); each wave owns a 32(M) x 64(N) macro-tile
//     -> 8 WMMA per K-step; clamp-free main loop + guarded tail
//   - per-chunk K-decode table in LDS: div/mod done once by 32 lanes
// ---------------------------------------------------------------------------

typedef __attribute__((ext_vector_type(16))) __bf16 v16bf;
typedef __attribute__((ext_vector_type(8)))  float  v8f;

struct GemmDesc {
    const float*          A;     // input tensor (implicit im2col source)
    const unsigned short* Wt;    // bf16 weights, K x N row-major (padded)
    const float*          bias;  // N floats or nullptr
    float*                out;   // M x N row-major
    int M, K, N;
    int mode;           // 0 = conv3d valid; 1 = conv3d-transpose; 2 = 3x3 SAME seq slice
    int relu;
    int accum;          // 1: acc initialized from out (z += conv)
    int D, H, Wd, C;    // input dims (mode2: H=rows, Wd=cols, C=Cin)
    int KH, SD, SH, OD, OH;
    int lsd, lsh;       // log2 strides (mode1 uses shift/mask)
    int T, t;           // mode2: sequence length and time index
};

// Async copy b32 global -> LDS, GVS addressing: SGPR base + 32-bit byte offset.
__device__ __forceinline__ void async_b32_to_lds(uint32_t lds_off, uint32_t byte_off,
                                                 const void* base) {
    asm volatile("global_load_async_to_lds_b32 %0, %1, %2"
                 :: "v"(lds_off), "v"(byte_off), "s"(base) : "memory");
}
__device__ __forceinline__ void wait_async0() {
    asm volatile("s_wait_asynccnt 0" ::: "memory");
}

__global__ __launch_bounds__(128) void wmma_gemm_kernel(GemmDesc g) {
    __shared__ __bf16 Bs[32][64];  // K x N weight tile (bf16)
    __shared__ int    toff[32];    // per-k precomputed offset (or ci for mode1)
    __shared__ int    tk0[32];     // kd (mode0/1) or kh (mode2)
    __shared__ int    tk1[32];     // kh (mode0/1) or kw (mode2)
    __shared__ int    tval[32];    // tail only: k < K

    const int tid  = threadIdx.x;
    const int lane = tid & 31;
    const int wave = tid >> 5;
    const int ln15 = lane & 15;
    const int hi   = (lane >= 16) ? 1 : 0;

    const int tileM = blockIdx.x * 4 + wave;   // 32 M-rows per wave
    const int n0    = blockIdx.y * 64;
    const int WdC   = g.Wd * g.C;
    const int mskD  = g.SD - 1, mskH = g.SH - 1;

    // ---- hoisted per-lane m-decode for the two M-subtiles ----
    bool mOKa[2]; int hha[2], wwca[2], oda[2], oha[2];
    const float* apa[2];
    #pragma unroll
    for (int u = 0; u < 2; ++u) {
        int mrow = tileM * 32 + u * 16 + ln15;
        mOKa[u] = (mrow < g.M);
        int mm = mOKa[u] ? mrow : 0;
        int baseA;
        if (g.mode == 2) {
            int ww = mm % g.Wd; int r2 = mm / g.Wd;
            int hh = r2 % g.H;  int n  = r2 / g.H;
            hha[u] = hh; wwca[u] = ww; oda[u] = 0; oha[u] = 0;
            baseA = (((n * g.T + g.t) * g.H + hh) * g.Wd + ww) * g.C;
        } else {
            int w  = mm % g.Wd; int r2 = mm / g.Wd;
            int oh = r2 % g.OH; int r3 = r2 / g.OH;
            int od = r3 % g.OD; int n  = r3 / g.OD;
            hha[u] = 0; wwca[u] = 0; oda[u] = od; oha[u] = oh;
            if (g.mode == 0)
                baseA = (((n * g.D + od * g.SD) * g.H + oh * g.SH) * g.Wd + w) * g.C;
            else
                baseA = n * g.D * g.H * WdC + w * g.C;   // + (id*H+ih)*WdC + ci
        }
        apa[u] = g.A + baseA;
    }

    // ---- loop-invariant staging offsets (bf16 tile: 8 b32 copies/thread) ----
    // dword d = tid + j*128 -> row kk = wave + 4j, dword-in-row = lane (2 cols)
    int colp = n0 + 2 * lane;                        // first of the column pair
    int colc = colp; if (colc > g.N - 2) colc = g.N - 2; if (colc < 0) colc = 0;
    const uint32_t colb = (uint32_t)colc * 2u;       // byte offset of pair
    const uint32_t N2   = (uint32_t)g.N * 2u;        // bf16 row pitch (bytes)
    uint32_t ldsoff[8], rowoff[8];
    #pragma unroll
    for (int j = 0; j < 8; ++j) {
        ldsoff[j] = (uint32_t)(uintptr_t)(&Bs[wave + 4 * j][2 * lane]);
        rowoff[j] = (uint32_t)(wave + 4 * j) * N2 + colb;
    }

    // ---- accumulators: 2 x 4 subtiles ----
    v8f acc[2][4];
    #pragma unroll
    for (int u = 0; u < 2; ++u)
        #pragma unroll
        for (int s = 0; s < 4; ++s) {
            if (g.accum) {
                #pragma unroll
                for (int r = 0; r < 8; ++r) {
                    int m  = tileM * 32 + u * 16 + r + hi * 8;
                    int co = n0 + s * 16 + ln15;
                    uint32_t o = (uint32_t)m * (uint32_t)g.N + (uint32_t)co;
                    acc[u][s][r] = (m < g.M && co < g.N) ? g.out[o] : 0.0f;
                }
            } else {
                #pragma unroll
                for (int r = 0; r < 8; ++r) acc[u][s][r] = 0.0f;
            }
        }

    const int kbh = hi << 4;

    // ======================= main loop: full 32-K tiles =======================
    const int ktMain = g.K >> 5;
    for (int kt = 0; kt < ktMain; ++kt) {
        const int k0 = kt * 32;
        const uint32_t k0b = (uint32_t)k0 * N2;
        __syncthreads();

        #pragma unroll
        for (int j = 0; j < 8; ++j)
            async_b32_to_lds(ldsoff[j], rowoff[j] + k0b, g.Wt);
        if (k0 + 32 < g.K)
            __builtin_prefetch((const char*)g.Wt + (size_t)(k0 + 32) * N2 + n0 * 2, 0, 1);

        if (tid < 32) {                 // K-decode (always valid in main loop)
            int k  = k0 + tid;
            int ci = k % g.C;
            int r  = k / g.C;
            if (g.mode == 2) {
                int kw = r % 3, kh = r / 3;
                toff[tid] = ((kh - 1) * g.Wd + (kw - 1)) * g.C + ci;
                tk0[tid] = kh; tk1[tid] = kw;
            } else {
                int kh = r % g.KH, kd = r / g.KH;
                toff[tid] = (g.mode == 0) ? (((kd * g.H + kh) * g.Wd) * g.C + ci) : ci;
                tk0[tid] = kd; tk1[tid] = kh;
            }
        }

        wait_async0();
        __syncthreads();

        v16bf afrag[2];
        #pragma unroll
        for (int u = 0; u < 2; ++u) {
            const float* ap = apa[u];
            const bool mOK  = mOKa[u];
            if (g.mode == 0) {
                #pragma unroll
                for (int e = 0; e < 16; ++e) {
                    int kk = ((e >> 3) << 4) + (hi << 3) + (e & 7);
                    float v = mOK ? ap[toff[kk]] : 0.0f;
                    afrag[u][e] = (__bf16)v;
                }
            } else if (g.mode == 2) {
                #pragma unroll
                for (int e = 0; e < 16; ++e) {
                    int kk = ((e >> 3) << 4) + (hi << 3) + (e & 7);
                    int ih = hha[u] + tk0[kk] - 1, iw = wwca[u] + tk1[kk] - 1;
                    bool ok = mOK && (ih >= 0) && (ih < g.H) && (iw >= 0) && (iw < g.Wd);
                    float v = ok ? ap[toff[kk]] : 0.0f;
                    afrag[u][e] = (__bf16)v;
                }
            } else {
                #pragma unroll
                for (int e = 0; e < 16; ++e) {
                    int kk = ((e >> 3) << 4) + (hi << 3) + (e & 7);
                    int dd = oda[u] - tk0[kk], h2 = oha[u] - tk1[kk];
                    int id = dd >> g.lsd, ih = h2 >> g.lsh;
                    bool ok = mOK && (dd >= 0) && (h2 >= 0) &&
                              ((dd & mskD) == 0) && ((h2 & mskH) == 0) &&
                              (id < g.D) && (ih < g.H);
                    float v = ok ? ap[id * g.H * WdC + ih * WdC + toff[kk]] : 0.0f;
                    afrag[u][e] = (__bf16)v;
                }
            }
        }

        #pragma unroll
        for (int s = 0; s < 4; ++s) {
            v16bf bfrag;
            #pragma unroll
            for (int e = 0; e < 16; ++e)
                bfrag[e] = Bs[((e >> 3) << 4) + (hi << 3) + (e & 7)][s * 16 + ln15];     // raw bf16, no convert
            acc[0][s] = __builtin_amdgcn_wmma_f32_16x16x32_bf16(
                            false, afrag[0], false, bfrag, (short)0, acc[0][s], false, false);
            acc[1][s] = __builtin_amdgcn_wmma_f32_16x16x32_bf16(
                            false, afrag[1], false, bfrag, (short)0, acc[1][s], false, false);
        }
    }

    // ======================= guarded tail (K % 32 != 0) =======================
    if (g.K & 31) {
        const int k0 = ktMain * 32;
        __syncthreads();

        #pragma unroll
        for (int j = 0; j < 8; ++j) {
            int krow = k0 + wave + 4 * j; if (krow > g.K - 1) krow = g.K - 1;
            async_b32_to_lds(ldsoff[j], (uint32_t)krow * N2 + colb, g.Wt);
        }

        if (tid < 32) {
            int k  = k0 + tid;
            int kv = (k < g.K) ? 1 : 0;
            int kc = kv ? k : 0;
            int ci = kc % g.C;
            int r  = kc / g.C;
            if (g.mode == 2) {
                int kw = r % 3, kh = r / 3;
                toff[tid] = ((kh - 1) * g.Wd + (kw - 1)) * g.C + ci;
                tk0[tid] = kh; tk1[tid] = kw;
            } else {
                int kh = r % g.KH, kd = r / g.KH;
                toff[tid] = (g.mode == 0) ? (((kd * g.H + kh) * g.Wd) * g.C + ci) : ci;
                tk0[tid] = kd; tk1[tid] = kh;
            }
            tval[tid] = kv;
        }

        wait_async0();
        __syncthreads();

        v16bf afrag[2];
        #pragma unroll
        for (int u = 0; u < 2; ++u) {
            const float* ap = apa[u];
            const bool mOK  = mOKa[u];
            if (g.mode == 0) {
                #pragma unroll
                for (int e = 0; e < 16; ++e) {
                    int kk = ((e >> 3) << 4) + (hi << 3) + (e & 7);
                    float v = (mOK && tval[kk]) ? ap[toff[kk]] : 0.0f;
                    afrag[u][e] = (__bf16)v;
                }
            } else if (g.mode == 2) {
                #pragma unroll
                for (int e = 0; e < 16; ++e) {
                    int kk = ((e >> 3) << 4) + (hi << 3) + (e & 7);
                    int ih = hha[u] + tk0[kk] - 1, iw = wwca[u] + tk1[kk] - 1;
                    bool ok = mOK && tval[kk] && (ih >= 0) && (ih < g.H) &&
                              (iw >= 0) && (iw < g.Wd);
                    float v = ok ? ap[toff[kk]] : 0.0f;
                    afrag[u][e] = (__bf16)v;
                }
            } else {
                #pragma unroll
                for (int e = 0; e < 16; ++e) {
                    int kk = ((e >> 3) << 4) + (hi << 3) + (e & 7);
                    int dd = oda[u] - tk0[kk], h2 = oha[u] - tk1[kk];
                    int id = dd >> g.lsd, ih = h2 >> g.lsh;
                    bool ok = mOK && tval[kk] && (dd >= 0) && (h2 >= 0) &&
                              ((dd & mskD) == 0) && ((h2 & mskH) == 0) &&
                              (id < g.D) && (ih < g.H);
                    float v = ok ? ap[id * g.H * WdC + ih * WdC + toff[kk]] : 0.0f;
                    afrag[u][e] = (__bf16)v;
                }
            }
        }

        const __bf16 bz = (__bf16)0.0f;
        #pragma unroll
        for (int s = 0; s < 4; ++s) {
            v16bf bfrag;
            #pragma unroll
            for (int e = 0; e < 16; ++e)
                bfrag[e] = (k0 + ((e >> 3) << 4) + (hi << 3) + (e & 7) < g.K) ? Bs[((e >> 3) << 4) + (hi << 3) + (e & 7)][s * 16 + ln15] : bz;
            acc[0][s] = __builtin_amdgcn_wmma_f32_16x16x32_bf16(
                            false, afrag[0], false, bfrag, (short)0, acc[0][s], false, false);
            acc[1][s] = __builtin_amdgcn_wmma_f32_16x16x32_bf16(
                            false, afrag[1], false, bfrag, (short)0, acc[1][s], false, false);
        }
    }

    // ---- epilogue: bias + relu + store (32-bit offsets) ----
    #pragma unroll
    for (int u = 0; u < 2; ++u)
        #pragma unroll
        for (int s = 0; s < 4; ++s) {
            int co = n0 + s * 16 + ln15;
            float bv = (!g.accum && g.bias != nullptr && co < g.N) ? g.bias[co] : 0.0f;
            #pragma unroll
            for (int r = 0; r < 8; ++r) {
                int m = tileM * 32 + u * 16 + r + hi * 8;
                if (m < g.M && co < g.N) {
                    float v = acc[u][s][r] + bv;
                    if (g.relu) v = fmaxf(v, 0.0f);
                    g.out[(uint32_t)m * (uint32_t)g.N + (uint32_t)co] = v;
                }
            }
        }
}

// ---------------------------------------------------------------------------
// Weight pre-conversion f32 -> bf16 (RNE via hardware convert)
// ---------------------------------------------------------------------------
__global__ void f32_to_bf16(const float* __restrict__ s, unsigned short* __restrict__ d,
                            int n) {
    int i = blockIdx.x * blockDim.x + threadIdx.x;
    if (i < n) {
        __bf16 b = (__bf16)s[i];
        unsigned short u; __builtin_memcpy(&u, &b, 2);
        d[i] = u;
    }
}

// ---------------------------------------------------------------------------
// ConvLSTM cell pointwise update (Keras gate order i,f,g,o; hard_sigmoid rec.)
// ---------------------------------------------------------------------------
__global__ void lstm_pointwise(const float* __restrict__ z,
                               float* __restrict__ c, float* __restrict__ h,
                               float* __restrict__ hseq,
                               int M, int lF, int T, int t, int HH, int WW) {
    int i = blockIdx.x * blockDim.x + threadIdx.x;
    int F = 1 << lF;
    if (i >= (M << lF)) return;
    int m = i >> lF, f = i & (F - 1);
    uint32_t zb = ((uint32_t)m << (lF + 2)) + (uint32_t)f;
    float zi = z[zb];
    float zf = z[zb + (uint32_t)F];
    float zg = z[zb + 2u * (uint32_t)F];
    float zo = z[zb + 3u * (uint32_t)F];
    auto hsg = [](float x) { return fminf(fmaxf(0.2f * x + 0.5f, 0.0f), 1.0f); };
    float cn = hsg(zf) * c[i] + hsg(zi) * tanhf(zg);
    float hn = hsg(zo) * tanhf(cn);
    c[i] = cn;
    h[i] = hn;
    int ww = m % WW; int r = m / WW; int hh = r % HH; int n = r / HH;
    uint32_t si = (uint32_t)((((n * T + t) * HH + hh) * WW + ww) << lF) + (uint32_t)f;
    hseq[si] = hn;
}

__global__ void fill_zero(float* p, size_t n) {
    size_t i = (size_t)blockIdx.x * blockDim.x + threadIdx.x;
    if (i < n) p[i] = 0.0f;
}

// ---------------------------------------------------------------------------
// Host orchestration
// ---------------------------------------------------------------------------
static inline size_t alignUp(size_t x, size_t a) { return (x + a - 1) & ~(a - 1); }
static inline int ilog2i(int x) { int l = 0; while ((1 << l) < x) ++l; return l; }

extern "C" void kernel_launch(void* const* d_in, const int* in_sizes, int n_in,
                              void* d_out, int out_size, void* d_ws, size_t ws_size,
                              hipStream_t stream) {
    (void)in_sizes; (void)n_in; (void)out_size; (void)ws_size;

    const float* x   = (const float*)d_in[0];
    const float* c1w = (const float*)d_in[1];
    const float* c1b = (const float*)d_in[2];
    const float* c2w = (const float*)d_in[3];
    const float* c2b = (const float*)d_in[4];
    const float* l1x = (const float*)d_in[5];
    const float* l1h = (const float*)d_in[6];
    const float* l1b = (const float*)d_in[7];
    const float* l2x = (const float*)d_in[8];
    const float* l2h = (const float*)d_in[9];
    const float* l2b = (const float*)d_in[10];
    const float* l3x = (const float*)d_in[11];
    const float* l3h = (const float*)d_in[12];
    const float* l3b = (const float*)d_in[13];
    const float* d1w = (const float*)d_in[14];
    const float* d1b = (const float*)d_in[15];
    const float* d2w = (const float*)d_in[16];
    const float* d2b = (const float*)d_in[17];
    float* out = (float*)d_out;

    const int B = 8;
    char* ws = (char*)d_ws;
    size_t off = 0;
    auto carveF = [&](size_t nf) -> float* {
        float* p = (float*)(ws + off);
        off = alignUp(off + nf * sizeof(float), 256);
        return p;
    };
    auto carveH = [&](size_t nh) -> unsigned short* {
        unsigned short* p = (unsigned short*)(ws + off);
        off = alignUp(off + (nh + 64) * sizeof(unsigned short), 256);  // +pad
        return p;
    };
    float* e1 = carveF((size_t)B * 55 * 55 * 10 * 128);  // conv1 out; reused for deconv1
    float* e2 = carveF((size_t)B * 26 * 26 * 10 * 64);   // conv2 out (LSTM input seq)
    float* h1 = carveF((size_t)B * 26 * 26 * 10 * 64);   // LSTM1 out seq
    float* h2 = carveF((size_t)B * 26 * 26 * 10 * 32);   // LSTM2 out seq
    float* h3 = carveF((size_t)B * 26 * 26 * 10 * 64);   // LSTM3 out seq
    float* zb = carveF((size_t)2080 * 256);              // gate pre-activations
    float* hb = carveF((size_t)2080 * 64);               // hidden state
    float* cb = carveF((size_t)2080 * 64);               // cell state

    // bf16 weight copies
    unsigned short* wc1 = carveH(121 * 128);
    unsigned short* wc2 = carveH(3200 * 64);
    unsigned short* w1x = carveH(576 * 256);
    unsigned short* w1h = carveH(576 * 256);
    unsigned short* w2x = carveH(576 * 128);
    unsigned short* w2h = carveH(288 * 128);
    unsigned short* w3x = carveH(288 * 256);
    unsigned short* w3h = carveH(576 * 256);
    unsigned short* wd1 = carveH(1600 * 128);
    unsigned short* wd2 = carveH(15488 * 1);

    auto conv_w = [&](const float* s, unsigned short* d, int n) {
        f32_to_bf16<<<dim3((unsigned)((n + 255) / 256)), dim3(256), 0, stream>>>(s, d, n);
    };
    conv_w(c1w, wc1, 121 * 128);
    conv_w(c2w, wc2, 3200 * 64);
    conv_w(l1x, w1x, 576 * 256);
    conv_w(l1h, w1h, 576 * 256);
    conv_w(l2x, w2x, 576 * 128);
    conv_w(l2h, w2h, 288 * 128);
    conv_w(l3x, w3x, 288 * 256);
    conv_w(l3h, w3h, 576 * 256);
    conv_w(d1w, wd1, 1600 * 128);
    conv_w(d2w, wd2, 15488 * 1);

    auto launch_gemm = [&](GemmDesc g) {
        g.lsd = ilog2i(g.SD);
        g.lsh = ilog2i(g.SH);
        int mt = (g.M + 31) / 32;      // 32 M-rows per wave
        int nt = (g.N + 63) / 64;
        dim3 grid((unsigned)((mt + 3) / 4), (unsigned)nt, 1);
        wmma_gemm_kernel<<<grid, dim3(128), 0, stream>>>(g);
    };

    // ---- encoder Conv3D 11x11 s4 (Cin=1 -> 128), ReLU ----
    {
        GemmDesc g{};
        g.A = x; g.Wt = wc1; g.bias = c1b; g.out = e1;
        g.M = B * 55 * 55 * 10; g.K = 11 * 11 * 1; g.N = 128;
        g.mode = 0; g.relu = 1; g.accum = 0;
        g.D = 227; g.H = 227; g.Wd = 10; g.C = 1;
        g.KH = 11; g.SD = 4; g.SH = 4; g.OD = 55; g.OH = 55; g.T = 1; g.t = 0;
        launch_gemm(g);
    }
    // ---- encoder Conv3D 5x5 s2 (128 -> 64), ReLU ----
    {
        GemmDesc g{};
        g.A = e1; g.Wt = wc2; g.bias = c2b; g.out = e2;
        g.M = B * 26 * 26 * 10; g.K = 5 * 5 * 128; g.N = 64;
        g.mode = 0; g.relu = 1; g.accum = 0;
        g.D = 55; g.H = 55; g.Wd = 10; g.C = 128;
        g.KH = 5; g.SD = 2; g.SH = 2; g.OD = 26; g.OH = 26; g.T = 1; g.t = 0;
        launch_gemm(g);
    }

    // ---- ConvLSTM2D stack ----
    auto run_lstm = [&](const float* xseq, int Cin, int F,
                        const unsigned short* wx, const unsigned short* wh,
                        const float* bias, float* hseq) {
        const int M = B * 26 * 10, T = 26, HH = 26, WW = 10;
        const int lF = ilog2i(F);
        size_t ns = (size_t)M * F;
        fill_zero<<<dim3((unsigned)((ns + 255) / 256)), dim3(256), 0, stream>>>(hb, ns);
        fill_zero<<<dim3((unsigned)((ns + 255) / 256)), dim3(256), 0, stream>>>(cb, ns);
        for (int t = 0; t < T; ++t) {
            GemmDesc gx{};
            gx.A = xseq; gx.Wt = wx; gx.bias = bias; gx.out = zb;
            gx.M = M; gx.K = 9 * Cin; gx.N = 4 * F;
            gx.mode = 2; gx.relu = 0; gx.accum = 0;
            gx.H = HH; gx.Wd = WW; gx.C = Cin; gx.SD = 1; gx.SH = 1;
            gx.T = 26; gx.t = t;
            launch_gemm(gx);

            GemmDesc gh{};
            gh.A = hb; gh.Wt = wh; gh.bias = nullptr; gh.out = zb;
            gh.M = M; gh.K = 9 * F; gh.N = 4 * F;
            gh.mode = 2; gh.relu = 0; gh.accum = 1;
            gh.H = HH; gh.Wd = WW; gh.C = F; gh.SD = 1; gh.SH = 1;
            gh.T = 1; gh.t = 0;
            launch_gemm(gh);

            lstm_pointwise<<<dim3((unsigned)((ns + 255) / 256)), dim3(256), 0, stream>>>(
                zb, cb, hb, hseq, M, lF, 26, t, HH, WW);
        }
    };
    run_lstm(e2, 64, 64, w1x, w1h, l1b, h1);
    run_lstm(h1, 64, 32, w2x, w2h, l2b, h2);
    run_lstm(h2, 32, 64, w3x, w3h, l3b, h3);

    // ---- decoder Conv3DTranspose 5x5 s2 (64 -> 128), ReLU (reuse e1) ----
    {
        GemmDesc g{};
        g.A = h3; g.Wt = wd1; g.bias = d1b; g.out = e1;
        g.M = B * 55 * 55 * 10; g.K = 5 * 5 * 64; g.N = 128;
        g.mode = 1; g.relu = 1; g.accum = 0;
        g.D = 26; g.H = 26; g.Wd = 10; g.C = 64;
        g.KH = 5; g.SD = 2; g.SH = 2; g.OD = 55; g.OH = 55; g.T = 1; g.t = 0;
        launch_gemm(g);
    }
    // ---- decoder Conv3DTranspose 11x11 s4 (128 -> 1), ReLU ----
    {
        GemmDesc g{};
        g.A = e1; g.Wt = wd2; g.bias = d2b; g.out = out;
        g.M = B * 227 * 227 * 10; g.K = 11 * 11 * 128; g.N = 1;
        g.mode = 1; g.relu = 1; g.accum = 0;
        g.D = 55; g.H = 55; g.Wd = 10; g.C = 128;
        g.KH = 11; g.SD = 4; g.SH = 4; g.OD = 227; g.OH = 227; g.T = 1; g.t = 0;
        launch_gemm(g);
    }
}
